// SimplicialCausalSelfAttention_24292335026388
// MI455X (gfx1250) — matrix operand order repaired and checked by reading, not compiled, not run on hardware
//
#include <hip/hip_runtime.h>
#include <hip/hip_bf16.h>

typedef __attribute__((ext_vector_type(16))) _Float16 v16h;
typedef __attribute__((ext_vector_type(8)))  _Float16 v8h;
typedef __attribute__((ext_vector_type(8)))  float    v8f;

static constexpr int Bb  = 2;
static constexpr int Tt  = 2048;
static constexpr int Cc  = 2048;
static constexpr int Hh  = 16;
static constexpr int HKk = 4;
static constexpr int HDd = 128;
static constexpr int SLD = 144;              // padded LDS row stride (halves); 288B rows keep 16B alignment
static constexpr float ATT_SCALE = 0.08838834764831845f; // 1/sqrt(128)

// ---------------- WMMA fragment helpers (CDNA5 16x16x32 f16 layouts) ----------------

__device__ __forceinline__ v8f wmma16(v16h a, v16h b, v8f c) {
    return __builtin_amdgcn_wmma_f32_16x16x32_f16(false, a, false, b, (short)0, c, false, false);
}

// A-matrix 16x32 (MxK), row-major source with leading dim ld.
// lanes 0-15: row=lane,   K in {0..15}
// lanes16-31: row=lane-16,K in {16..31}
__device__ __forceinline__ v16h load_a_frag(const _Float16* base, int ld) {
    int lane  = threadIdx.x & 31;
    int row   = lane & 15;
    int khalf = (lane >> 4) * 16;
    const _Float16* p = base + row * ld;
    v16h a;
#pragma unroll
    for (int e = 0; e < 16; ++e) a[e] = p[khalf + e];
    return a;
}

// B-matrix 32x16 where the SOURCE is stored transposed (N x K row-major, ld = K stride),
// i.e. b[e] = src[col*ld + ksel + e] -> 32 contiguous bytes per lane (2x ds_load_b128).
__device__ __forceinline__ v16h load_bT_frag(const _Float16* base, int ld) {
    int lane = threadIdx.x & 31;
    int col  = lane & 15;
    int ksel = (lane >> 4) * 16;
    const _Float16* p = base + col * ld + ksel;
    v16h b;
#pragma unroll
    for (int e = 0; e < 16; ++e) b[e] = p[e];
    return b;
}

// stage rows x 128 f16 tile (global, row stride srcld) into LDS (ld=SLD), 16B vector copies
__device__ __forceinline__ void stage_tile_f16(_Float16* dst, const _Float16* src, int srcld) {
    for (int i = threadIdx.x; i < 128 * 16; i += 256) {
        int row = i >> 4;
        int c8  = (i & 15) * 8;
        *(v8h*)(dst + row * SLD + c8) = *(const v8h*)(src + (size_t)row * srcld + c8);
    }
}

// stage 128x128 f16 tile TRANSPOSED: global [row][col] (row stride srcld) -> LDS [col][row].
// Contiguous 16B global reads; strided LDS element stores (one-time cost, makes all
// subsequent B-fragment reads contiguous).
__device__ __forceinline__ void stage_tile_f16_T(_Float16* dst, const _Float16* src, int srcld) {
    for (int i = threadIdx.x; i < 128 * 16; i += 256) {
        int row = i >> 4;
        int c0  = (i & 15) * 8;
        v8h v = *(const v8h*)(src + (size_t)row * srcld + c0);
#pragma unroll
        for (int e = 0; e < 8; ++e) dst[(c0 + e) * SLD + row] = v[e];
    }
}

// ---------------- Kernel 1/5: tiled GEMM, C(f32) = A * Bw, f16 WMMA ----------------
// Block tile 128x128, K-step 32, 256 threads = 8 waves tiled 4(M) x 2(N):
// each wave owns a 32x64 sub-tile -> 2 A-frags + 4 B-frags per k-step, 8 WMMAs.
template <typename AT>
__global__ __launch_bounds__(256) void gemm_f16w(const AT* __restrict__ A,
                                                 const float* __restrict__ Bw,
                                                 float* __restrict__ Co,
                                                 int M, int N, int K) {
    (void)M;
    constexpr int ALD  = 40;  // 32 + pad
    constexpr int BLDT = 40;  // transposed B tile: 128 rows(n) x 32(k) + pad
    __shared__ _Float16 As[128 * ALD];
    __shared__ _Float16 BsT[128 * BLDT];

    const int tid  = threadIdx.x;
    const int wave = tid >> 5;
    const int wm   = wave & 3;   // M tile index (rows 32*wm)
    const int wn   = wave >> 2;  // N tile index (cols 64*wn)
    const int lane = tid & 31;
    const int half = lane >> 4;
    const int col  = lane & 15;
    const int m0 = blockIdx.y * 128;
    const int n0 = blockIdx.x * 128;

    v8f acc[2][4];
#pragma unroll
    for (int mi = 0; mi < 2; ++mi)
#pragma unroll
        for (int ni = 0; ni < 4; ++ni)
#pragma unroll
            for (int e = 0; e < 8; ++e) acc[mi][ni][e] = 0.0f;

    for (int k0 = 0; k0 < K; k0 += 32) {
        // stage A 128x32 (f32 or f16 -> f16), row-major
        {
            int row = tid >> 1, cb = (tid & 1) * 16;
            const AT* src = A + (size_t)(m0 + row) * K + k0 + cb;
#pragma unroll
            for (int e = 0; e < 16; ++e) As[row * ALD + cb + e] = (_Float16)src[e];
            if (k0 + 32 < K) __builtin_prefetch(src + 32, 0, 0);
        }
        // stage B 32x128 (f32 -> f16) TRANSPOSED to [n][k]
        {
            int row = tid >> 3, cb = (tid & 7) * 16;
            const float* src = Bw + (size_t)(k0 + row) * N + n0 + cb;
#pragma unroll
            for (int e = 0; e < 16; ++e) BsT[(cb + e) * BLDT + row] = (_Float16)src[e];
            if (k0 + 32 < K) __builtin_prefetch(src + (size_t)32 * N, 0, 0);
        }
        __syncthreads();

        v16h a0 = load_a_frag(As + (32 * wm) * ALD, ALD);
        v16h a1 = load_a_frag(As + (32 * wm + 16) * ALD, ALD);
#pragma unroll
        for (int ni = 0; ni < 4; ++ni) {
            v16h b = load_bT_frag(BsT + (64 * wn + ni * 16) * BLDT, BLDT);
            acc[0][ni] = wmma16(a0, b, acc[0][ni]);
            acc[1][ni] = wmma16(a1, b, acc[1][ni]);
        }
        __syncthreads();
    }

#pragma unroll
    for (int mi = 0; mi < 2; ++mi)
#pragma unroll
        for (int ni = 0; ni < 4; ++ni)
#pragma unroll
            for (int r = 0; r < 8; ++r) {
                int gm = m0 + 32 * wm + 16 * mi + r + 8 * half;
                int gn = n0 + 64 * wn + ni * 16 + col;
                Co[(size_t)gm * N + gn] = acc[mi][ni][r];
            }
}

// ---------------- Kernel 2: RoPE + RMS + head-major f16 pack ----------------
__global__ __launch_bounds__(256) void rope_rms_pack(const float* __restrict__ qraw,
                                                     const float* __restrict__ kraw,
                                                     const float* __restrict__ vraw,
                                                     const float* __restrict__ cosp,
                                                     const float* __restrict__ sinp,
                                                     _Float16* __restrict__ q16,
                                                     _Float16* __restrict__ k16,
                                                     _Float16* __restrict__ v16) {
    const int bt = blockIdx.x;
    const int b  = bt / Tt;
    const int t  = bt % Tt;
    const int wave = threadIdx.x >> 5;
    const int lane = threadIdx.x & 31;
    const float* cs = cosp + t * 64;
    const float* sn = sinp + t * 64;

    for (int u = wave; u < Hh + HKk; u += 8) {
        const float* src;
        _Float16* dst;
        if (u < Hh) {
            src = qraw + (size_t)bt * Cc + u * HDd;
            dst = q16 + ((size_t)(b * Hh + u) * Tt + t) * HDd;
        } else {
            int hk = u - Hh;
            src = kraw + (size_t)bt * (HKk * HDd) + hk * HDd;
            dst = k16 + ((size_t)(b * HKk + hk) * Tt + t) * HDd;
        }
        float o[4];
        int d0 = lane * 2, d1 = lane * 2 + 1;
        float x1a = src[d0], x2a = src[d0 + 64], ca = cs[d0], sa = sn[d0];
        float x1b = src[d1], x2b = src[d1 + 64], cb = cs[d1], sb = sn[d1];
        o[0] = x1a * ca - x2a * sa;  // pos d0
        o[1] = x1a * sa + x2a * ca;  // pos d0+64
        o[2] = x1b * cb - x2b * sb;  // pos d1
        o[3] = x1b * sb + x2b * cb;  // pos d1+64
        float ss = o[0]*o[0] + o[1]*o[1] + o[2]*o[2] + o[3]*o[3];
#pragma unroll
        for (int m = 1; m < 32; m <<= 1) ss += __shfl_xor(ss, m, 32);
        float rs = rsqrtf(ss * (1.0f / HDd) + 1e-6f);
        dst[d0]      = (_Float16)(o[0] * rs);
        dst[d0 + 64] = (_Float16)(o[1] * rs);
        dst[d1]      = (_Float16)(o[2] * rs);
        dst[d1 + 64] = (_Float16)(o[3] * rs);
    }
    // V: straight f32 -> f16 repack
    for (int i = threadIdx.x; i < HKk * HDd; i += 256) {
        int hk = i >> 7, d = i & 127;
        v16[((size_t)(b * HKk + hk) * Tt + t) * HDd + d] =
            (_Float16)vraw[(size_t)bt * (HKk * HDd) + i];
    }
}

// ---------------- Kernel 3: flash pass A -> y1, m, l ----------------
__global__ __launch_bounds__(256) void attn_pass1(const _Float16* __restrict__ q16,
                                                  const _Float16* __restrict__ k16,
                                                  const _Float16* __restrict__ v16,
                                                  float* __restrict__ y1f,
                                                  _Float16* __restrict__ y1h,
                                                  float* __restrict__ mbuf,
                                                  float* __restrict__ lbuf) {
    extern __shared__ _Float16 smem[];
    _Float16* Qs  = smem;                 // [qrow][kd]  row-major
    _Float16* Ks  = smem + 128 * SLD;     // [key][kd]   row-major (B^T source for S)
    _Float16* VsT = smem + 2 * 128 * SLD; // [d][key]    transposed (B^T source for P@V)
    _Float16* Ps  = smem + 3 * 128 * SLD; // [qrow][key] row-major (A source for P@V)

    const int qb = blockIdx.x, h = blockIdx.y, b = blockIdx.z;
    const int hk = h / (Hh / HKk);
    const int wave = threadIdx.x >> 5;
    const int lane = threadIdx.x & 31;
    const int half = lane >> 4;
    const int col  = lane & 15;

    const _Float16* qbase = q16 + ((size_t)(b * Hh + h) * Tt + qb * 128) * HDd;
    const _Float16* kbase = k16 + (size_t)(b * HKk + hk) * Tt * HDd;
    const _Float16* vbase = v16 + (size_t)(b * HKk + hk) * Tt * HDd;

    stage_tile_f16(Qs, qbase, HDd);

    float m_run[8], l_run[8];
    v8f Oacc[8];
#pragma unroll
    for (int r = 0; r < 8; ++r) { m_run[r] = -__builtin_inff(); l_run[r] = 0.0f; }
#pragma unroll
    for (int dt = 0; dt < 8; ++dt)
#pragma unroll
        for (int r = 0; r < 8; ++r) Oacc[dt][r] = 0.0f;

    for (int jb = 0; jb <= qb; ++jb) {
        __syncthreads();
        stage_tile_f16(Ks, kbase + (size_t)jb * 128 * HDd, HDd);
        stage_tile_f16_T(VsT, vbase + (size_t)jb * 128 * HDd, HDd);
        if (jb < qb) {  // prefetch next K/V blocks: 256 threads x 128B = whole 32KB tile
            __builtin_prefetch((const char*)(kbase + (size_t)(jb + 1) * 128 * HDd) + threadIdx.x * 128, 0, 0);
            __builtin_prefetch((const char*)(vbase + (size_t)(jb + 1) * 128 * HDd) + threadIdx.x * 128, 0, 0);
        }
        __syncthreads();

        // S = Q * K^T  (contiguous B^T fragment loads from Ks)
        v8f s[8];
#pragma unroll
        for (int nt = 0; nt < 8; ++nt)
#pragma unroll
            for (int r = 0; r < 8; ++r) s[nt][r] = 0.0f;
#pragma unroll
        for (int kk = 0; kk < 4; ++kk) {
            v16h a = load_a_frag(Qs + (16 * wave) * SLD + kk * 32, SLD);
#pragma unroll
            for (int nt = 0; nt < 8; ++nt) {
                v16h bf = load_bT_frag(Ks + (nt * 16) * SLD + kk * 32, SLD);
                s[nt] = wmma16(a, bf, s[nt]);
            }
        }

        // scale + causal mask + online softmax (row stats live in a 16-lane half)
        float mnew[8];
#pragma unroll
        for (int r = 0; r < 8; ++r) mnew[r] = m_run[r];
#pragma unroll
        for (int nt = 0; nt < 8; ++nt) {
            int kcol = jb * 128 + nt * 16 + col;
#pragma unroll
            for (int r = 0; r < 8; ++r) {
                int qrow = qb * 128 + 16 * wave + r + 8 * half;
                float sv = s[nt][r] * ATT_SCALE;
                if (kcol > qrow) sv = -__builtin_inff();
                s[nt][r] = sv;
                mnew[r] = fmaxf(mnew[r], sv);
            }
        }
#pragma unroll
        for (int m = 1; m < 16; m <<= 1)
#pragma unroll
            for (int r = 0; r < 8; ++r) mnew[r] = fmaxf(mnew[r], __shfl_xor(mnew[r], m, 32));

        float alpha[8], rowsum[8];
#pragma unroll
        for (int r = 0; r < 8; ++r) {
            alpha[r] = __expf(m_run[r] - mnew[r]);
            m_run[r] = mnew[r];
            rowsum[r] = 0.0f;
        }
#pragma unroll
        for (int nt = 0; nt < 8; ++nt)
#pragma unroll
            for (int r = 0; r < 8; ++r) {
                float p = __expf(s[nt][r] - mnew[r]);  // exp(-inf)=0 handles mask
                rowsum[r] += p;
                Ps[(16 * wave + r + 8 * half) * SLD + nt * 16 + col] = (_Float16)p;
            }
#pragma unroll
        for (int m = 1; m < 16; m <<= 1)
#pragma unroll
            for (int r = 0; r < 8; ++r) rowsum[r] += __shfl_xor(rowsum[r], m, 32);
#pragma unroll
        for (int r = 0; r < 8; ++r) l_run[r] = l_run[r] * alpha[r] + rowsum[r];
#pragma unroll
        for (int dt = 0; dt < 8; ++dt)
#pragma unroll
            for (int r = 0; r < 8; ++r) Oacc[dt][r] *= alpha[r];
        __syncthreads();

        // O += P * V  (contiguous B^T fragment loads from VsT)
#pragma unroll
        for (int kk = 0; kk < 4; ++kk) {
            v16h a = load_a_frag(Ps + (16 * wave) * SLD + kk * 32, SLD);
#pragma unroll
            for (int dt = 0; dt < 8; ++dt) {
                v16h bf = load_bT_frag(VsT + (dt * 16) * SLD + kk * 32, SLD);
                Oacc[dt] = wmma16(a, bf, Oacc[dt]);
            }
        }
    }
    __syncthreads();

    // y1 = O / l ; store f32 + f16 and row stats
    const size_t obase = ((size_t)(b * Hh + h) * Tt + qb * 128) * HDd;
#pragma unroll
    for (int dt = 0; dt < 8; ++dt)
#pragma unroll
        for (int r = 0; r < 8; ++r) {
            int row = 16 * wave + r + 8 * half;
            int d   = dt * 16 + col;
            float y = Oacc[dt][r] / l_run[r];
            y1f[obase + (size_t)row * HDd + d] = y;
            y1h[obase + (size_t)row * HDd + d] = (_Float16)y;
        }
    if (col == 0) {
        size_t sb = (size_t)(b * Hh + h) * Tt + qb * 128;
#pragma unroll
        for (int r = 0; r < 8; ++r) {
            int row = 16 * wave + r + 8 * half;
            mbuf[sb + row] = m_run[r];
            lbuf[sb + row] = l_run[r];
        }
    }
}

// ---------------- Kernel 4: pass B -> y2 = att @ y1 ; y = mix1*y1 + mix2*y2 ----------------
__global__ __launch_bounds__(256) void attn_pass2(const _Float16* __restrict__ q16,
                                                  const _Float16* __restrict__ k16,
                                                  const _Float16* __restrict__ y1h,
                                                  const float* __restrict__ y1f,
                                                  const float* __restrict__ mbuf,
                                                  const float* __restrict__ lbuf,
                                                  const float* __restrict__ mix1p,
                                                  const float* __restrict__ mix2p,
                                                  _Float16* __restrict__ yc16) {
    extern __shared__ _Float16 smem[];
    _Float16* Qs  = smem;
    _Float16* Ks  = smem + 128 * SLD;
    _Float16* YsT = smem + 2 * 128 * SLD;  // y1 tile transposed [d][key]
    _Float16* Ps  = smem + 3 * 128 * SLD;

    const int qb = blockIdx.x, h = blockIdx.y, b = blockIdx.z;
    const int hk = h / (Hh / HKk);
    const int wave = threadIdx.x >> 5;
    const int lane = threadIdx.x & 31;
    const int half = lane >> 4;
    const int col  = lane & 15;
    const float mix1 = mix1p[0], mix2 = mix2p[0];

    const _Float16* qbase = q16 + ((size_t)(b * Hh + h) * Tt + qb * 128) * HDd;
    const _Float16* kbase = k16 + (size_t)(b * HKk + hk) * Tt * HDd;
    const _Float16* ybase = y1h + (size_t)(b * Hh + h) * Tt * HDd;

    stage_tile_f16(Qs, qbase, HDd);

    float m_i[8], invl[8];
    {
        size_t sb = (size_t)(b * Hh + h) * Tt + qb * 128;
#pragma unroll
        for (int r = 0; r < 8; ++r) {
            int row = 16 * wave + r + 8 * half;
            m_i[r]  = mbuf[sb + row];
            invl[r] = 1.0f / lbuf[sb + row];
        }
    }

    v8f Oacc[8];
#pragma unroll
    for (int dt = 0; dt < 8; ++dt)
#pragma unroll
        for (int r = 0; r < 8; ++r) Oacc[dt][r] = 0.0f;

    for (int jb = 0; jb <= qb; ++jb) {
        __syncthreads();
        stage_tile_f16(Ks, kbase + (size_t)jb * 128 * HDd, HDd);
        stage_tile_f16_T(YsT, ybase + (size_t)jb * 128 * HDd, HDd);
        if (jb < qb) {
            __builtin_prefetch((const char*)(kbase + (size_t)(jb + 1) * 128 * HDd) + threadIdx.x * 128, 0, 0);
            __builtin_prefetch((const char*)(ybase + (size_t)(jb + 1) * 128 * HDd) + threadIdx.x * 128, 0, 0);
        }
        __syncthreads();

        v8f s[8];
#pragma unroll
        for (int nt = 0; nt < 8; ++nt)
#pragma unroll
            for (int r = 0; r < 8; ++r) s[nt][r] = 0.0f;
#pragma unroll
        for (int kk = 0; kk < 4; ++kk) {
            v16h a = load_a_frag(Qs + (16 * wave) * SLD + kk * 32, SLD);
#pragma unroll
            for (int nt = 0; nt < 8; ++nt) {
                v16h bf = load_bT_frag(Ks + (nt * 16) * SLD + kk * 32, SLD);
                s[nt] = wmma16(a, bf, s[nt]);
            }
        }

        // att = exp(s*scale - m) / l  (exact reconstruction from pass-A stats)
#pragma unroll
        for (int nt = 0; nt < 8; ++nt) {
            int kcol = jb * 128 + nt * 16 + col;
#pragma unroll
            for (int r = 0; r < 8; ++r) {
                int qrow = qb * 128 + 16 * wave + r + 8 * half;
                float p = (kcol > qrow) ? 0.0f
                                        : __expf(s[nt][r] * ATT_SCALE - m_i[r]) * invl[r];
                Ps[(16 * wave + r + 8 * half) * SLD + nt * 16 + col] = (_Float16)p;
            }
        }
        __syncthreads();

#pragma unroll
        for (int kk = 0; kk < 4; ++kk) {
            v16h a = load_a_frag(Ps + (16 * wave) * SLD + kk * 32, SLD);
#pragma unroll
            for (int dt = 0; dt < 8; ++dt) {
                v16h bf = load_bT_frag(YsT + (dt * 16) * SLD + kk * 32, SLD);
                Oacc[dt] = wmma16(a, bf, Oacc[dt]);
            }
        }
    }
    __syncthreads();

    // y = mix1*y1 + mix2*y2, repacked to (B, T, C) f16 for the output projection
    const size_t y1base = ((size_t)(b * Hh + h) * Tt + qb * 128) * HDd;
#pragma unroll
    for (int dt = 0; dt < 8; ++dt)
#pragma unroll
        for (int r = 0; r < 8; ++r) {
            int row = 16 * wave + r + 8 * half;   // local query row
            int d   = dt * 16 + col;
            float y1v = y1f[y1base + (size_t)row * HDd + d];
            float y   = mix1 * y1v + mix2 * Oacc[dt][r];
            int t = qb * 128 + row;
            yc16[((size_t)(b * Tt + t)) * Cc + h * HDd + d] = (_Float16)y;
        }
}

// ---------------- launch ----------------
extern "C" void kernel_launch(void* const* d_in, const int* in_sizes, int n_in,
                              void* d_out, int out_size, void* d_ws, size_t ws_size,
                              hipStream_t stream) {
    (void)in_sizes; (void)n_in; (void)out_size; (void)ws_size;
    const float* x     = (const float*)d_in[0];
    const float* cosp  = (const float*)d_in[1];
    const float* sinp  = (const float*)d_in[2];
    const float* Wq    = (const float*)d_in[3];
    const float* Wk    = (const float*)d_in[4];
    const float* Wv    = (const float*)d_in[5];
    const float* Wproj = (const float*)d_in[6];
    const float* mix1  = (const float*)d_in[7];
    const float* mix2  = (const float*)d_in[8];

    char* w = (char*)d_ws;
    auto alloc = [&](size_t bytes) {
        char* p = w;
        w += (bytes + 255) & ~(size_t)255;
        return p;
    };
    const size_t M = (size_t)Bb * Tt;  // 4096
    float*    qraw = (float*)alloc(M * Cc * 4);
    float*    kraw = (float*)alloc(M * HKk * HDd * 4);
    float*    vraw = (float*)alloc(M * HKk * HDd * 4);
    _Float16* q16  = (_Float16*)alloc((size_t)Bb * Hh * Tt * HDd * 2);
    _Float16* k16  = (_Float16*)alloc((size_t)Bb * HKk * Tt * HDd * 2);
    _Float16* v16  = (_Float16*)alloc((size_t)Bb * HKk * Tt * HDd * 2);
    float*    mbuf = (float*)alloc((size_t)Bb * Hh * Tt * 4);
    float*    lbuf = (float*)alloc((size_t)Bb * Hh * Tt * 4);
    float*    y1f  = (float*)alloc((size_t)Bb * Hh * Tt * HDd * 4);
    _Float16* y1h  = (_Float16*)alloc((size_t)Bb * Hh * Tt * HDd * 2);
    _Float16* yc16 = (_Float16*)alloc(M * Cc * 2);

    // 1) QKV projections
    gemm_f16w<float><<<dim3(Cc / 128, M / 128), 256, 0, stream>>>(x, Wq, qraw, (int)M, Cc, Cc);
    gemm_f16w<float><<<dim3((HKk * HDd) / 128, M / 128), 256, 0, stream>>>(x, Wk, kraw, (int)M, HKk * HDd, Cc);
    gemm_f16w<float><<<dim3((HKk * HDd) / 128, M / 128), 256, 0, stream>>>(x, Wv, vraw, (int)M, HKk * HDd, Cc);

    // 2) RoPE + RMS + pack
    rope_rms_pack<<<Bb * Tt, 256, 0, stream>>>(qraw, kraw, vraw, cosp, sinp, q16, k16, v16);

    // 3/4) two-hop causal attention
    size_t smem = (size_t)4 * 128 * SLD * sizeof(_Float16);  // ~144 KB of 320 KB LDS
    attn_pass1<<<dim3(Tt / 128, Hh, Bb), 256, smem, stream>>>(q16, k16, v16, y1f, y1h, mbuf, lbuf);
    attn_pass2<<<dim3(Tt / 128, Hh, Bb), 256, smem, stream>>>(q16, k16, y1h, y1f, mbuf, lbuf,
                                                              mix1, mix2, yc16);

    // 5) output projection -> f32
    gemm_f16w<_Float16><<<dim3(Cc / 128, M / 128), 256, 0, stream>>>(yc16, Wproj, (float*)d_out,
                                                                     (int)M, Cc, Cc);
}
